// FP8Linear_30451318128859
// MI455X (gfx1250) — matrix operand order repaired and checked by reading, not compiled, not run on hardware
//
#include <hip/hip_runtime.h>
#include <hip/hip_bf16.h>

// ---------------------------------------------------------------------------
// FP8 (bf16-emulated) block-quantized linear:  out = Xdeq @ Wdeq^T + bias
//   Kernel 1: per-row activation quant  -> Xq (bf16, ws) + x_scale (f32, ws)
//   Kernel 2: bf16 WMMA GEMM; B (weights) staged global->LDS with
//             double-buffered global_load_async_to_lds_b128 (ASYNCcnt),
//             A direct from global; folded block scales + bias epilogue.
// ---------------------------------------------------------------------------

#define FP8_MAX 448.0f
#define QBS 128   // weight quant block size (reference block_size)

// LDS B tile: 128 rows (n) x 128 cols (k) bf16, padded row stride to kill
// bank conflicts: 136 elems = 272 B -> per-row bank offset of 4.
#define B_STRIDE     136
#define B_BUF_BYTES  (128 * B_STRIDE * 2)   // 34816 B per buffer
#define LDS_BYTES    (2 * B_BUF_BYTES)      // 69632 B double buffered

typedef __attribute__((ext_vector_type(16))) __bf16 v16bf;
typedef __attribute__((ext_vector_type(8)))  __bf16 v8bf;
typedef __attribute__((ext_vector_type(8)))  float  v8f;

union Frag16 {
    v16bf v;
    v8bf  h[2];
};

// ---------------------------------------------------------------------------
// Kernel 1: one 256-thread block per row of X.
//   amax = max(|X[row,:]|, eps); scale = amax/448; Xq = bf16(X/scale)
// ---------------------------------------------------------------------------
__global__ __launch_bounds__(256)
void rowquant_kernel(const float* __restrict__ X,
                     __hip_bfloat16* __restrict__ Xq,
                     float* __restrict__ xs,
                     int K)
{
    const int row = blockIdx.x;
    const float* xr = X + (size_t)row * K;

    float amax = 0.0f;
    for (int i = threadIdx.x; i < K; i += 256)
        amax = fmaxf(amax, fabsf(xr[i]));

    // wave32 reduction
    #pragma unroll
    for (int off = 16; off > 0; off >>= 1)
        amax = fmaxf(amax, __shfl_xor(amax, off, 32));

    __shared__ float red[8];
    __shared__ float samax;
    const int wv = threadIdx.x >> 5;
    if ((threadIdx.x & 31) == 0) red[wv] = amax;
    __syncthreads();
    if (threadIdx.x == 0) {
        float m = red[0];
        #pragma unroll
        for (int i = 1; i < 8; ++i) m = fmaxf(m, red[i]);
        samax = fmaxf(m, 1e-12f);
    }
    __syncthreads();

    const float scale = samax * (1.0f / FP8_MAX);
    const float inv   = FP8_MAX / samax;

    __hip_bfloat16* qr = Xq + (size_t)row * K;
    for (int i = threadIdx.x; i < K; i += 256) {
        float q = xr[i] * inv;
        q = fminf(fmaxf(q, -FP8_MAX), FP8_MAX);
        qr[i] = __float2bfloat16(q);
    }
    if (threadIdx.x == 0) xs[row] = scale;
}

// ---------------------------------------------------------------------------
// Kernel 2: GEMM  out[m,n] = xs[m] * sum_kb wscale[nb,kb] *
//                            (sum_{k in kb} Xq[m,k]*Wq[n,k])  + bias[n]
// WG tile 128(M) x 128(N), 8 waves; each wave: 2x4 fragments of 16x16.
// B tile (128x128 bf16) staged via async global->LDS, double buffered.
// ---------------------------------------------------------------------------
extern __shared__ char smem[];

__global__ __launch_bounds__(256)
void gemm_bf16_wmma_kernel(const __hip_bfloat16* __restrict__ Xq,
                           const __hip_bfloat16* __restrict__ Wq,
                           const float* __restrict__ xs,
                           const float* __restrict__ wscale,
                           const float* __restrict__ bias,
                           float* __restrict__ out,
                           int M, int N, int K)
{
    const int lane     = threadIdx.x & 31;
    const int wv       = threadIdx.x >> 5;
    const int waveM    = wv & 3;          // 4 wave rows  -> 4*32  = 128 M
    const int waveN    = wv >> 2;         // 2 wave cols  -> 2*64  = 128 N
    const int nb       = blockIdx.x;      // N block (also weight scale row)
    const int mb       = blockIdx.y;
    const int mBase    = mb * 128 + waveM * 32;
    const int nTile    = nb * 128;
    const int KB       = K / QBS;

    const int laneHalf = lane >> 4;       // 0: lanes 0-15, 1: lanes 16-31
    const int lane16   = lane & 15;

    const __bf16* A = (const __bf16*)Xq;

    // A (16x32, MxK) fragment layout: lane16 = row; 8-elem chunks at
    // k + laneHalf*8 and k + laneHalf*8 + 16.
    const __bf16* aRow[2];
    #pragma unroll
    for (int mi = 0; mi < 2; ++mi)
        aRow[mi] = A + (size_t)(mBase + mi * 16 + lane16) * K + laneHalf * 8;

    // ---- async B-tile copy: 128x128 bf16 = 2048 x 16B chunks, 8/thread ----
    const unsigned ldsBase = (unsigned)(uintptr_t)smem;   // LDS byte offset
    const char*    wbase   = (const char*)Wq;             // uniform 64-bit base

    auto issue_b_tile = [&](int buf, int kb) {
        const unsigned lb = ldsBase + (unsigned)(buf * B_BUF_BYTES);
        const int k0 = kb * QBS;
        #pragma unroll
        for (int i = 0; i < 8; ++i) {
            const int c  = (int)threadIdx.x + i * 256;
            const int n  = c >> 4;          // row within tile
            const int kc = c & 15;          // 16B chunk within row
            const unsigned loff = lb + (unsigned)(n * (B_STRIDE * 2) + kc * 16);
            const unsigned goff =
                (unsigned)((((size_t)(nTile + n)) * K + (size_t)k0 + kc * 8) * 2);
            asm volatile("global_load_async_to_lds_b128 %0, %1, %2"
                         :: "v"(loff), "v"(goff), "s"(wbase)
                         : "memory");
        }
    };

    // B fragment base inside an LDS buffer: row = waveN*64 + ni*16 + lane16,
    // 8-elem K chunks at kk + laneHalf*8 and kk + laneHalf*8 + 16 (two 16B ds_load chunks).
    const unsigned bFragBase[4] = {
        (unsigned)((waveN * 64 +  0 + lane16) * (B_STRIDE * 2) + laneHalf * 16),
        (unsigned)((waveN * 64 + 16 + lane16) * (B_STRIDE * 2) + laneHalf * 16),
        (unsigned)((waveN * 64 + 32 + lane16) * (B_STRIDE * 2) + laneHalf * 16),
        (unsigned)((waveN * 64 + 48 + lane16) * (B_STRIDE * 2) + laneHalf * 16),
    };

    v8f acc[2][4] = {};

    issue_b_tile(0, 0);   // prologue fill of buffer 0

    for (int kb = 0; kb < KB; ++kb) {
        const int cur = kb & 1;
        const int k0  = kb * QBS;

        if (kb + 1 < KB) {
            issue_b_tile(cur ^ 1, kb + 1);           // 8 more in flight
            asm volatile("s_wait_asynccnt 0x8" ::: "memory");  // cur done
        } else {
            asm volatile("s_wait_asynccnt 0x0" ::: "memory");
        }
        __syncthreads();                             // cur visible to all waves

        const float s = wscale[nb * KB + kb];        // uniform -> scalar load
        const char* bbuf = smem + cur * B_BUF_BYTES;

        // Prefetch next K-chunk of A while WMMAs grind this one.
        if (kb + 1 < KB) {
            __builtin_prefetch(aRow[0] + k0 + QBS, 0, 0);
            __builtin_prefetch(aRow[1] + k0 + QBS, 0, 0);
        }

        v8f bacc[2][4] = {};

        #pragma unroll
        for (int kk = 0; kk < QBS; kk += 32) {
            const int k = k0 + kk;

            Frag16 a[2];
            #pragma unroll
            for (int mi = 0; mi < 2; ++mi) {
                a[mi].h[0] = *(const v8bf*)(aRow[mi] + k);
                a[mi].h[1] = *(const v8bf*)(aRow[mi] + k + 16);
            }
            Frag16 b[4];
            #pragma unroll
            for (int ni = 0; ni < 4; ++ni) {
                const char* bp = bbuf + bFragBase[ni] + kk * 2;
                b[ni].h[0] = *(const v8bf*)(bp);
                b[ni].h[1] = *(const v8bf*)(bp + 32);
            }

            #pragma unroll
            for (int mi = 0; mi < 2; ++mi)
                #pragma unroll
                for (int ni = 0; ni < 4; ++ni)
                    bacc[mi][ni] = __builtin_amdgcn_wmma_f32_16x16x32_bf16(
                        false, a[mi].v, false, b[ni].v,
                        (short)0, bacc[mi][ni], false, false);
        }

        // Fold the (uniform) weight block scale into the running accumulator.
        #pragma unroll
        for (int mi = 0; mi < 2; ++mi)
            #pragma unroll
            for (int ni = 0; ni < 4; ++ni)
                #pragma unroll
                for (int j = 0; j < 8; ++j)
                    acc[mi][ni][j] += s * bacc[mi][ni][j];

        __syncthreads();   // all waves done reading cur before it is refilled
    }

    // Epilogue: C/D layout -> VGPR j holds row (base + j + laneHalf*8), col lane16.
    #pragma unroll
    for (int mi = 0; mi < 2; ++mi) {
        const int rbase = mBase + mi * 16 + laneHalf * 8;
        float xsv[8];
        #pragma unroll
        for (int j = 0; j < 8; ++j) xsv[j] = xs[rbase + j];

        #pragma unroll
        for (int ni = 0; ni < 4; ++ni) {
            const int c = nTile + waveN * 64 + ni * 16 + lane16;
            const float bv = bias[c];
            #pragma unroll
            for (int j = 0; j < 8; ++j)
                out[(size_t)(rbase + j) * N + c] = xsv[j] * acc[mi][ni][j] + bv;
        }
    }
}

// ---------------------------------------------------------------------------
// Host-side launcher
// ---------------------------------------------------------------------------
extern "C" void kernel_launch(void* const* d_in, const int* in_sizes, int n_in,
                              void* d_out, int out_size, void* d_ws, size_t ws_size,
                              hipStream_t stream)
{
    const float*          X      = (const float*)d_in[0];
    const __hip_bfloat16* Wq     = (const __hip_bfloat16*)d_in[1];
    const float*          Wscale = (const float*)d_in[2];
    const float*          bias   = (const float*)d_in[3];
    // d_in[4] = block_size scalar (=128, baked in as QBS)

    const int N = in_sizes[3];                       // bias length
    const int K = (int)((long long)in_sizes[1] / N); // weight is [N, K]
    const int M = (int)((long long)in_sizes[0] / K); // X flattened to [M, K]

    // Workspace layout: [ Xq bf16 : M*K ][ x_scale f32 : M ]
    __hip_bfloat16* Xq = (__hip_bfloat16*)d_ws;
    float*          xs = (float*)((char*)d_ws + (size_t)M * K * sizeof(__hip_bfloat16));

    rowquant_kernel<<<M, 256, 0, stream>>>(X, Xq, xs, K);

    dim3 grid(N / 128, M / 128);
    gemm_bf16_wmma_kernel<<<grid, 256, LDS_BYTES, stream>>>(Xq, Wq, xs, Wscale, bias,
                                                            (float*)d_out, M, N, K);
}
